// SelfAttention1D_50551765073979
// MI455X (gfx1250) — matrix operand order repaired and checked by reading, not compiled, not run on hardware
//
#include <hip/hip_runtime.h>
#include <stdint.h>

typedef __attribute__((ext_vector_type(16))) __bf16 v16bf;
typedef __attribute__((ext_vector_type(8)))  __bf16 v8bf;
typedef __attribute__((ext_vector_type(8)))  float  v8f;
typedef __attribute__((ext_vector_type(4)))  int    v4i;

#define B_  8
#define C_  256
#define L_  2048

#if defined(__gfx1250__) && __has_builtin(__builtin_amdgcn_global_load_async_to_lds_b128)
#define HAVE_ASYNC_LDS 1
#else
#define HAVE_ASYNC_LDS 0
#endif

static __device__ __forceinline__ __bf16 f2bf(float f) {
    union { float f; uint32_t u; } a; a.f = f;
    uint32_t r = (a.u + 0x7fffu + ((a.u >> 16) & 1u)) >> 16;   // RNE
    union { uint16_t s; __bf16 b; } o; o.s = (uint16_t)r;
    return o.b;
}

union V16U { v16bf v; v8bf h[2]; };

// ---------------- Stage 0: f32 -> bf16 weight conversion ----------------
__global__ void cvt_f32_bf16(const float* __restrict__ s, __bf16* __restrict__ d, int n) {
    int i = blockIdx.x * blockDim.x + threadIdx.x;
    if (i < n) d[i] = f2bf(s[i]);
}

// ---------------- Stage 1: QKV projection (WMMA bf16) -------------------
// Q -> Qt[b][l][c], K -> Kt[b][l][c] (transposed), V -> Vc[b][c][l] (natural)
__global__ __launch_bounds__(256)
void qkv_kernel(const float* __restrict__ x,
                const __bf16* __restrict__ Wq, const __bf16* __restrict__ Wk,
                const __bf16* __restrict__ Wv,
                const float* __restrict__ bq, const float* __restrict__ bk,
                const float* __restrict__ bv,
                __bf16* __restrict__ Qt, __bf16* __restrict__ Kt, __bf16* __restrict__ Vc)
{
    __shared__ __bf16 xt[128][40];   // x tile transposed [l][c'], 80B rows (16B-aligned)

    const int bid = blockIdx.x;
    const int mat = bid >> 8;            // 0=Q 1=K 2=V (256 blocks each)
    const int rem = bid & 255;
    const int b   = rem >> 5;            // batch
    const int oh  = (rem >> 4) & 1;      // output-channel half
    const int l0  = (rem & 15) * 128;    // l tile
    const int tid  = threadIdx.x;
    const int wv   = tid >> 5, lane = tid & 31;
    const int lh   = lane >> 4, n = lane & 15;
    const int obase = oh * 128 + wv * 16;

    const __bf16* W    = (mat == 0) ? Wq : (mat == 1) ? Wk : Wv;
    const float*  bias = (mat == 0) ? bq : (mat == 1) ? bk : bv;

    v8f acc[8] = {};

    for (int kk = 0; kk < C_; kk += 32) {
        // cooperative staged transpose with packed b32 LDS stores:
        // thread handles c'-pair (2*rp, 2*rp+1), 8 l-values each
        {
            const int rp = tid >> 4;           // 0..15
            const int ls = (tid & 15) * 8;     // l local base
            const float* s0 = x + (((size_t)b * C_ + kk + 2 * rp) * L_ + l0 + ls);
            const float* s1 = s0 + L_;
            #pragma unroll
            for (int q = 0; q < 8; ++q) {
                union { __bf16 h[2]; uint32_t u; } p;
                p.h[0] = f2bf(s0[q]);
                p.h[1] = f2bf(s1[q]);
                *(uint32_t*)&xt[ls + q][2 * rp] = p.u;
            }
        }
        __syncthreads();

        // A fragment: W row (obase+n), chunks at k = 8*lh and 16+8*lh (ISA A-map)
        V16U a;
        const __bf16* wrow = W + (size_t)(obase + n) * C_ + kk;
        a.h[0] = *(const v8bf*)(wrow + 8 * lh);
        a.h[1] = *(const v8bf*)(wrow + 16 + 8 * lh);

        #pragma unroll
        for (int s = 0; s < 8; ++s) {
            // B fragment: row l = s*16+n, chunks at k = 8*lh and 16+8*lh (ISA B-map)
            V16U bf;
            bf.h[0] = *(const v8bf*)&xt[s * 16 + n][8 * lh];
            bf.h[1] = *(const v8bf*)&xt[s * 16 + n][16 + 8 * lh];
            acc[s] = __builtin_amdgcn_wmma_f32_16x16x32_bf16(
                         false, a.v, false, bf.v, (short)0, acc[s], false, false);
        }
        __syncthreads();
    }

    // bias + store in attention-friendly layouts
    #pragma unroll
    for (int s = 0; s < 8; ++s) {
        const int l = l0 + s * 16 + n;
        #pragma unroll
        for (int r = 0; r < 8; ++r) {
            const int o = obase + r + 8 * lh;            // D-map: m = r + 8*lh
            const __bf16 v = f2bf(acc[s][r] + bias[o]);
            if (mat == 2) {
                Vc[((size_t)b * C_ + o) * L_ + l] = v;
            } else {
                __bf16* dst = (mat == 0) ? Qt : Kt;
                dst[((size_t)b * L_ + l) * C_ + o] = v;
            }
        }
    }
}

// ---------------- Stage 2: flash attention (WMMA bf16) ------------------
__global__ __launch_bounds__(256)
void attn_kernel(const float* __restrict__ x,
                 const __bf16* __restrict__ Qt, const __bf16* __restrict__ Kt,
                 const __bf16* __restrict__ Vc, float* __restrict__ out)
{
    __shared__ __bf16 plds[8][16][32];     // per-wave P tile (D-layout -> A-layout relay)
#if HAVE_ASYNC_LDS
    // double-buffered K tile [32 keys][256 ch], rows padded to 272 elems
    // (544B = 34 banks stride: conflict-free, 16B-aligned)
    __shared__ __bf16 kbuf[2][32][272];
#endif

    const int bid = blockIdx.x;
    const int b   = bid >> 4;
    const int it  = bid & 15;
    const int tid = threadIdx.x;
    const int wv  = tid >> 5, lane = tid & 31;
    const int lh  = lane >> 4, n = lane & 15;
    const int ibase = it * 128 + wv * 16;  // 16 query rows per wave

    const __bf16* Qb = Qt + (size_t)b * L_ * C_;
    const __bf16* Kb = Kt + (size_t)b * L_ * C_;
    const __bf16* Vb = Vc + (size_t)b * C_ * L_;

#if HAVE_ASYNC_LDS
    // each thread DMAs 4 x 16B chunks of the 16KB K tile (coalesced across lanes)
    auto stage_k = [&](int jt, int bufi) {
        #pragma unroll
        for (int q = 0; q < 4; ++q) {
            const int chunk = q * 256 + tid;       // 0..1023
            const int krow  = chunk >> 5;          // 0..31
            const int kcol  = chunk & 31;          // 16B chunk within row
            const __bf16* g = Kb + (size_t)(jt + krow) * C_ + kcol * 8;
            __builtin_amdgcn_global_load_async_to_lds_b128(
                (__attribute__((address_space(1))) v4i*)(void*)g,
                (__attribute__((address_space(3))) v4i*)&kbuf[bufi][krow][kcol * 8],
                0, 0);
        }
    };
#endif

    // preload Q A-fragments (row i = ibase + (lane&15), reused for all 64 key tiles)
    V16U qa[8];
    #pragma unroll
    for (int kc = 0; kc < 8; ++kc) {
        const __bf16* qrow = Qb + (size_t)(ibase + n) * C_ + kc * 32;
        qa[kc].h[0] = *(const v8bf*)(qrow + 16 * lh);
        qa[kc].h[1] = *(const v8bf*)(qrow + 16 * lh + 8);
    }

    v8f o[16] = {};                // O accumulators: 16 rows x 256 channels
    float rmax[8], rsum[8];
    #pragma unroll
    for (int r = 0; r < 8; ++r) { rmax[r] = -3.0e38f; rsum[r] = 0.0f; }

    const float scale = 0.0625f;   // 1/sqrt(256)

#if HAVE_ASYNC_LDS
    int cur = 0;
    stage_k(0, 0);
    asm volatile("s_wait_asynccnt 0" ::: "memory");
    __syncthreads();
#endif

    for (int j0 = 0; j0 < L_; j0 += 32) {
#if HAVE_ASYNC_LDS
        if (j0 + 32 < L_) stage_k(j0 + 32, cur ^ 1);   // overlap DMA with compute
#endif
        // ---- S = Q^T K over 256 channels (two 16-wide key halves) ----
        v8f s0 = {}, s1 = {};
        #pragma unroll
        for (int kc = 0; kc < 8; ++kc) {
#if HAVE_ASYNC_LDS
            const __bf16* k0 = &kbuf[cur][n][kc * 32 + 16 * lh];
            const __bf16* k1 = &kbuf[cur][16 + n][kc * 32 + 16 * lh];
#else
            const __bf16* k0 = Kb + (size_t)(j0 + n)      * C_ + kc * 32 + 16 * lh;
            const __bf16* k1 = Kb + (size_t)(j0 + 16 + n) * C_ + kc * 32 + 16 * lh;
#endif
            V16U kb0, kb1;
            kb0.h[0] = *(const v8bf*)k0; kb0.h[1] = *(const v8bf*)(k0 + 8);
            kb1.h[0] = *(const v8bf*)k1; kb1.h[1] = *(const v8bf*)(k1 + 8);
            s0 = __builtin_amdgcn_wmma_f32_16x16x32_bf16(false, qa[kc].v, false, kb0.v,
                                                         (short)0, s0, false, false);
            s1 = __builtin_amdgcn_wmma_f32_16x16x32_bf16(false, qa[kc].v, false, kb1.v,
                                                         (short)0, s1, false, false);
        }
#if !HAVE_ASYNC_LDS
        if (j0 + 32 < L_)
            __builtin_prefetch(Kb + (size_t)(j0 + 32 + n) * C_, 0, 1);
#endif

        // ---- online softmax over this 32-key chunk ----
        #pragma unroll
        for (int r = 0; r < 8; ++r) {
            float a0 = s0[r] * scale, a1 = s1[r] * scale;
            float cm = fmaxf(a0, a1);
            #pragma unroll
            for (int off = 1; off < 16; off <<= 1)          // stays within 16-lane half
                cm = fmaxf(cm, __shfl_xor(cm, off, 32));
            float mnew = fmaxf(rmax[r], cm);
            float corr = __expf(rmax[r] - mnew);
            rmax[r] = mnew;
            float p0 = __expf(a0 - mnew);
            float p1 = __expf(a1 - mnew);
            float cs = p0 + p1;
            #pragma unroll
            for (int off = 1; off < 16; off <<= 1)
                cs += __shfl_xor(cs, off, 32);
            rsum[r] = rsum[r] * corr + cs;
            #pragma unroll
            for (int t = 0; t < 16; ++t) o[t][r] *= corr;   // rescale O rows (same m-map)
            plds[wv][r + 8 * lh][n]      = f2bf(p0);        // D-layout write
            plds[wv][r + 8 * lh][16 + n] = f2bf(p1);
        }

        // P back out in A-layout (wave-private region; DS ops in-order per wave)
        V16U pa;
        pa.h[0] = *(const v8bf*)&plds[wv][n][16 * lh];
        pa.h[1] = *(const v8bf*)&plds[wv][n][16 * lh + 8];

        // ---- O += P * V over all 16 channel tiles ----
        #pragma unroll
        for (int t = 0; t < 16; ++t) {
            const __bf16* vrow = Vb + (size_t)(t * 16 + n) * L_ + j0 + 16 * lh;
            V16U vb;
            vb.h[0] = *(const v8bf*)vrow;
            vb.h[1] = *(const v8bf*)(vrow + 8);
            o[t] = __builtin_amdgcn_wmma_f32_16x16x32_bf16(false, pa.v, false, vb.v,
                                                           (short)0, o[t], false, false);
        }

#if HAVE_ASYNC_LDS
        asm volatile("s_wait_asynccnt 0" ::: "memory");  // next tile landed
        __syncthreads();                                  // all waves done with cur
        cur ^= 1;
#endif
    }

    // ---- epilogue: out = O * (1/rowsum) + x (residual) ----
    float inv[8];
    #pragma unroll
    for (int r = 0; r < 8; ++r) inv[r] = 1.0f / rsum[r];
    #pragma unroll
    for (int t = 0; t < 16; ++t) {
        const int c = t * 16 + n;
        #pragma unroll
        for (int r = 0; r < 8; ++r) {
            const int i = ibase + r + 8 * lh;
            const size_t idx = ((size_t)b * C_ + c) * L_ + i;
            out[idx] = o[t][r] * inv[r] + x[idx];
        }
    }
}

// ------------------------------ launcher --------------------------------
extern "C" void kernel_launch(void* const* d_in, const int* in_sizes, int n_in,
                              void* d_out, int out_size, void* d_ws, size_t ws_size,
                              hipStream_t stream)
{
    const float* x  = (const float*)d_in[0];
    const float* Wq = (const float*)d_in[1];
    const float* bq = (const float*)d_in[2];
    const float* Wk = (const float*)d_in[3];
    const float* bk = (const float*)d_in[4];
    const float* Wv = (const float*)d_in[5];
    const float* bv = (const float*)d_in[6];
    float* out = (float*)d_out;

    char* ws = (char*)d_ws;
    const size_t WSZ = (size_t)C_ * C_ * 2;          // 128 KB per weight matrix
    const size_t MSZ = (size_t)B_ * L_ * C_ * 2;     // 8 MB per Q/K/V tensor
    __bf16* Wqb = (__bf16*)(ws);
    __bf16* Wkb = (__bf16*)(ws + WSZ);
    __bf16* Wvb = (__bf16*)(ws + 2 * WSZ);
    __bf16* Qt  = (__bf16*)(ws + 3 * WSZ);
    __bf16* Kt  = (__bf16*)(ws + 3 * WSZ + MSZ);
    __bf16* Vc  = (__bf16*)(ws + 3 * WSZ + 2 * MSZ);

    cvt_f32_bf16<<<(C_ * C_ + 255) / 256, 256, 0, stream>>>(Wq, Wqb, C_ * C_);
    cvt_f32_bf16<<<(C_ * C_ + 255) / 256, 256, 0, stream>>>(Wk, Wkb, C_ * C_);
    cvt_f32_bf16<<<(C_ * C_ + 255) / 256, 256, 0, stream>>>(Wv, Wvb, C_ * C_);

    // 3 matrices * 8 batches * 2 channel-halves * 16 l-tiles = 768 blocks
    qkv_kernel<<<768, 256, 0, stream>>>(x, Wqb, Wkb, Wvb, bq, bk, bv, Qt, Kt, Vc);

    // 8 batches * 16 query-tiles(128) = 128 blocks, 8 waves each
    attn_kernel<<<128, 256, 0, stream>>>(x, Qt, Kt, Vc, out);
}
